// INLDiTAttention_25580825215323
// MI455X (gfx1250) — matrix operand order repaired and checked by reading, not compiled, not run on hardware
//
#include <hip/hip_runtime.h>

#define D_MODEL 1024
#define NUM_HEADS 16
#define NUM_KV 4
#define HEAD_DIM 64
#define KV_DIM 256
#define BATCH 2
#define SEQ 2048
#define MROWS (BATCH * SEQ)

#define AS1 __attribute__((address_space(1)))
#define AS3 __attribute__((address_space(3)))

typedef __attribute__((ext_vector_type(16))) __bf16 bfrag;   // 16 bf16 = 32B (A/B operand, 8 VGPRs)
typedef __attribute__((ext_vector_type(8)))  __bf16 bhalf8;  // 8 bf16 = 16B
typedef __attribute__((ext_vector_type(8)))  float  v8f;     // C/D accumulator (8 VGPRs)
typedef __attribute__((ext_vector_type(4)))  int    v4i;     // 16B payload for async b128

// ---------------------------------------------------------------- helpers ---

__device__ __forceinline__ __bf16 to_bf16(float f) {
  unsigned u = __builtin_bit_cast(unsigned, f);
  unsigned r = (u + 0x7FFFu + ((u >> 16) & 1u)) >> 16;   // round-to-nearest-even
  unsigned short s = (unsigned short)r;
  return __builtin_bit_cast(__bf16, s);
}

__device__ __forceinline__ bfrag join8(bhalf8 lo, bhalf8 hi) {
  bfrag r;
#pragma unroll
  for (int i = 0; i < 8; ++i) { r[i] = lo[i]; r[i + 8] = hi[i]; }
  return r;
}

// A-matrix 16x32 bf16 fragment (ISA 7.12.2): lanes 0-15 hold M=lane,
// K={0..7,16..23}; lanes 16-31 hold M=lane-16, K={8..15,24..31}.
__device__ __forceinline__ bfrag load_frag_a(const __bf16* base, int stride) {
  int lane = threadIdx.x & 31;
  int m    = lane & 15;
  int koff = (lane >> 4) << 3;               // 0 or 8
  const __bf16* p = base + m * stride + koff;
  return join8(*(const bhalf8*)p, *(const bhalf8*)(p + 16));
}

// B-matrix 32x16 bf16 fragment: lanes 0-15 hold N=lane, K={0..7,16..23};
// lanes 16-31 hold N=lane-16, K={8..15,24..31}.
__device__ __forceinline__ bfrag load_frag_b(const __bf16* base, int stride) {
  int lane = threadIdx.x & 31;
  int koff = (lane >> 4) << 3;
  const __bf16* p = base + (lane & 15);
  bfrag r;
#pragma unroll
  for (int i = 0; i < 8; ++i) {
    r[i]     = p[(koff + i) * stride];
    r[i + 8] = p[(16 + koff + i) * stride];
  }
  return r;
}

__device__ __forceinline__ v8f wmma_bf16(bfrag a, bfrag b, v8f c) {
  return __builtin_amdgcn_wmma_f32_16x16x32_bf16(false, a, false, b, (short)0, c,
                                                 false, false);
}

// Async memory -> LDS copy of 16 bytes per lane (GLOBAL_LOAD_ASYNC_TO_LDS_B128,
// tracked by ASYNCcnt).  Fallback: synchronous VGPR round-trip.
__device__ __forceinline__ void async_cp16(const __bf16* g, __bf16* l) {
#if __has_builtin(__builtin_amdgcn_global_load_async_to_lds_b128)
  __builtin_amdgcn_global_load_async_to_lds_b128((AS1 v4i*)g, (AS3 v4i*)l, 0, 0);
#else
  *(bhalf8*)l = *(const bhalf8*)g;
#endif
}

__device__ __forceinline__ void wait_async0() {
#if __has_builtin(__builtin_amdgcn_s_wait_asynccnt)
  __builtin_amdgcn_s_wait_asynccnt(0);
#else
  asm volatile("s_wait_asynccnt 0x0" ::: "memory");
#endif
}

// ------------------------------------------------------------ fp32 -> bf16 --

__global__ void f32_to_bf16_kernel(const float* __restrict__ in,
                                   __bf16* __restrict__ out, long n) {
  long i = (long)blockIdx.x * blockDim.x + threadIdx.x;
  if (i < n) out[i] = to_bf16(in[i]);
}

// ------------------------------------------------------------------- GEMM ---
// C[M,N] (fp32) = A[M,K] (bf16, row major) * B[K,N] (bf16, row major)
// block = 256 threads (8 waves), tile 128(M) x 64(N), K step 32.
// Double-buffered: async-prefetch tile i+1 into LDS while WMMAs consume tile i.

__global__ __launch_bounds__(256) void gemm_bf16(const __bf16* __restrict__ A,
                                                 const __bf16* __restrict__ Bm,
                                                 float* __restrict__ C,
                                                 int M, int N, int K) {
  __shared__ __attribute__((aligned(16))) __bf16 As[2][128][40];  // 80B rows
  __shared__ __attribute__((aligned(16))) __bf16 Bs[2][32][72];   // 144B rows

  const int tid  = threadIdx.x;
  const int wave = tid >> 5;
  const long bm  = (long)blockIdx.x * 128;
  const int  bn  = blockIdx.y * 64;

  // per-thread staging slices: A 128x32 -> 2 x 16B/thread; B 32x64 -> 1 x 16B/thread
  const int ra = tid >> 1, ca = (tid & 1) * 16;
  const int rb = tid >> 3, cb = (tid & 7) * 8;

  auto stage = [&](int k0, int buf) {
    const __bf16* ga = A + (bm + ra) * (long)K + k0 + ca;
    async_cp16(ga,     &As[buf][ra][ca]);
    async_cp16(ga + 8, &As[buf][ra][ca + 8]);
    const __bf16* gb = Bm + (long)(k0 + rb) * N + bn + cb;
    async_cp16(gb, &Bs[buf][rb][cb]);
  };

  v8f acc[4];
#pragma unroll
  for (int i = 0; i < 4; ++i) {
    v8f z = {0.f, 0.f, 0.f, 0.f, 0.f, 0.f, 0.f, 0.f};
    acc[i] = z;
  }

  const int niter = K >> 5;
  stage(0, 0);
  for (int it = 0; it < niter; ++it) {
    wait_async0();        // this wave's slice of tile `it` has landed in LDS
    __syncthreads();      // everyone's slices landed; previous readers done
    if (it + 1 < niter) stage((it + 1) << 5, (it + 1) & 1);
    const int buf = it & 1;
    bfrag a = load_frag_a(&As[buf][wave * 16][0], 40);
#pragma unroll
    for (int nf = 0; nf < 4; ++nf) {
      bfrag bfr = load_frag_b(&Bs[buf][0][nf * 16], 72);
      acc[nf] = wmma_bf16(a, bfr, acc[nf]);
    }
    // no trailing barrier: wmma's s_wait_dscnt means LDS reads of buf are
    // complete before this wave reaches the next top-of-loop barrier.
  }

  const int lane = tid & 31;
  const int cn   = lane & 15;
  const long wm  = bm + wave * 16 + ((lane >> 4) << 3);
#pragma unroll
  for (int nf = 0; nf < 4; ++nf) {
    int col = bn + nf * 16 + cn;
#pragma unroll
    for (int r = 0; r < 8; ++r) C[(wm + r) * (long)N + col] = acc[nf][r];
  }
}

// ------------------------------------------------------------------- RoPE ---
// T: fp32 (rows, heads*64). out[j] = (t_even*c - t_odd*s)*scale,
// out[32+j] = (t_even*s + t_odd*c)*scale  (split-halves variant).

__global__ void rope_bf16(const float* __restrict__ T,
                          const float* __restrict__ cosT,
                          const float* __restrict__ sinT,
                          __bf16* __restrict__ out,
                          int rows, int heads, float scale) {
  long i = (long)blockIdx.x * blockDim.x + threadIdx.x;
  long total = (long)rows * heads * 32;
  if (i >= total) return;
  int  j   = (int)(i & 31);
  long t   = i >> 5;
  int  h   = (int)(t % heads);
  long row = t / heads;
  int  pos = (int)(row % SEQ);
  float c = cosT[pos * 32 + j];
  float s = sinT[pos * 32 + j];
  const float* p = T + (row * heads + h) * HEAD_DIM;
  float t1 = p[2 * j], t2 = p[2 * j + 1];
  __bf16* q = out + (row * heads + h) * HEAD_DIM;
  q[j]      = to_bf16((t1 * c - t2 * s) * scale);
  q[32 + j] = to_bf16((t1 * s + t2 * c) * scale);
}

// -------------------------------------------------------- flash attention ---
// Grid: (SEQ/128, NUM_HEADS, BATCH). 8 waves; each wave owns 16 query rows.
// 1/sqrt(HEAD_DIM) scale is pre-folded into Q. Double-buffered K/V tiles;
// V staged with async-to-LDS, K staged with a transpose scatter.

__global__ __launch_bounds__(256) void attn_fa(const __bf16* __restrict__ Qb,
                                               const __bf16* __restrict__ Kb,
                                               const __bf16* __restrict__ Vb,
                                               __bf16* __restrict__ Ob) {
  __shared__ __attribute__((aligned(16))) __bf16 Kt[2][HEAD_DIM][40];  // [d][key]
  __shared__ __attribute__((aligned(16))) __bf16 Vt[2][32][72];        // [key][d]
  __shared__ __attribute__((aligned(16))) __bf16 Ps[8][16][40];        // per-wave P

  const int tid  = threadIdx.x;
  const int wave = tid >> 5;
  const int lane = tid & 31;
  const int cl   = lane & 15;
  const int hlf  = lane >> 4;
  const int h    = blockIdx.y;
  const int b    = blockIdx.z;
  const int kvh  = h >> 2;   // NUM_HEADS / NUM_KV = 4

  const long rowQ = (long)b * SEQ + (long)blockIdx.x * 128 + wave * 16;
  const __bf16* qbase = Qb + rowQ * D_MODEL + h * HEAD_DIM;
  bfrag qf0 = load_frag_a(qbase, D_MODEL);        // d = 0..31
  bfrag qf1 = load_frag_a(qbase + 32, D_MODEL);   // d = 32..63

  const int kk = tid >> 3;
  const int d0 = (tid & 7) << 3;
  auto stage = [&](int kb, int buf) {
    long krow = (long)b * SEQ + kb * 32 + kk;
    const __bf16* kp = Kb + krow * KV_DIM + kvh * HEAD_DIM + d0;
    bhalf8 kv8 = *(const bhalf8*)kp;               // transpose scatter (sync)
#pragma unroll
    for (int i = 0; i < 8; ++i) Kt[buf][d0 + i][kk] = kv8[i];
    async_cp16(Vb + krow * KV_DIM + kvh * HEAD_DIM + d0, &Vt[buf][kk][d0]);
  };

  v8f oacc[4];
#pragma unroll
  for (int i = 0; i < 4; ++i) {
    v8f z = {0.f, 0.f, 0.f, 0.f, 0.f, 0.f, 0.f, 0.f};
    oacc[i] = z;
  }
  float mrun[8], lrun[8];
#pragma unroll
  for (int r = 0; r < 8; ++r) { mrun[r] = -1e30f; lrun[r] = 0.f; }

  const int nkb = SEQ / 32;
  stage(0, 0);
  for (int kb = 0; kb < nkb; ++kb) {
    wait_async0();
    __syncthreads();
    if (kb + 1 < nkb) stage(kb + 1, (kb + 1) & 1);
    const int buf = kb & 1;

    // S = Q * K^T  (16 queries x 32 keys), two 16-key column fragments
    v8f s0 = {0.f, 0.f, 0.f, 0.f, 0.f, 0.f, 0.f, 0.f};
    v8f s1 = {0.f, 0.f, 0.f, 0.f, 0.f, 0.f, 0.f, 0.f};
    s0 = wmma_bf16(qf0, load_frag_b(&Kt[buf][0][0], 40), s0);
    s0 = wmma_bf16(qf1, load_frag_b(&Kt[buf][32][0], 40), s0);
    s1 = wmma_bf16(qf0, load_frag_b(&Kt[buf][0][16], 40), s1);
    s1 = wmma_bf16(qf1, load_frag_b(&Kt[buf][32][16], 40), s1);

    // online softmax; C-layout: VGPR r + 8*hlf = row, 16-lane halves = columns
    float alpha[8];
#pragma unroll
    for (int r = 0; r < 8; ++r) {
      float m = fmaxf(s0[r], s1[r]);
#pragma unroll
      for (int off = 8; off > 0; off >>= 1) m = fmaxf(m, __shfl_xor(m, off, 32));
      float mn = fmaxf(mrun[r], m);
      float p0 = __expf(s0[r] - mn);
      float p1 = __expf(s1[r] - mn);
      s0[r] = p0; s1[r] = p1;
      float rs = p0 + p1;
#pragma unroll
      for (int off = 8; off > 0; off >>= 1) rs += __shfl_xor(rs, off, 32);
      float al = __expf(mrun[r] - mn);
      lrun[r]  = lrun[r] * al + rs;
      mrun[r]  = mn;
      alpha[r] = al;
    }
#pragma unroll
    for (int df = 0; df < 4; ++df)
#pragma unroll
      for (int r = 0; r < 8; ++r) oacc[df][r] *= alpha[r];

    // P: C-layout -> LDS -> A-layout for the P*V WMMA (wave-private staging)
#pragma unroll
    for (int r = 0; r < 8; ++r) {
      Ps[wave][hlf * 8 + r][cl]      = to_bf16(s0[r]);
      Ps[wave][hlf * 8 + r][16 + cl] = to_bf16(s1[r]);
    }
    bfrag pa = load_frag_a(&Ps[wave][0][0], 40);
#pragma unroll
    for (int df = 0; df < 4; ++df)
      oacc[df] = wmma_bf16(pa, load_frag_b(&Vt[buf][0][df * 16], 72), oacc[df]);
  }

  float inv[8];
#pragma unroll
  for (int r = 0; r < 8; ++r) inv[r] = 1.0f / lrun[r];
#pragma unroll
  for (int df = 0; df < 4; ++df) {
    int col = h * HEAD_DIM + df * 16 + cl;
#pragma unroll
    for (int r = 0; r < 8; ++r) {
      long row = rowQ + hlf * 8 + r;
      Ob[row * D_MODEL + col] = to_bf16(oacc[df][r] * inv[r]);
    }
  }
}

// ----------------------------------------------------------------- launch ---

extern "C" void kernel_launch(void* const* d_in, const int* in_sizes, int n_in,
                              void* d_out, int out_size, void* d_ws, size_t ws_size,
                              hipStream_t stream) {
  const float* x    = (const float*)d_in[0];
  const float* cosT = (const float*)d_in[1];
  const float* sinT = (const float*)d_in[2];
  const float* Wq   = (const float*)d_in[3];
  const float* Wk   = (const float*)d_in[4];
  const float* Wv   = (const float*)d_in[5];
  const float* Wo   = (const float*)d_in[6];
  float* out = (float*)d_out;

  char* w = (char*)d_ws;
  size_t off = 0;
  auto take = [&](size_t bytes) -> char* {
    char* p = w + off;
    off += (bytes + 255) & ~(size_t)255;
    return p;
  };
  __bf16* xb  = (__bf16*)take((size_t)MROWS * D_MODEL * 2);
  __bf16* wqb = (__bf16*)take((size_t)D_MODEL * D_MODEL * 2);
  __bf16* wkb = (__bf16*)take((size_t)D_MODEL * KV_DIM * 2);
  __bf16* wvb = (__bf16*)take((size_t)D_MODEL * KV_DIM * 2);
  __bf16* wob = (__bf16*)take((size_t)D_MODEL * D_MODEL * 2);
  float*  qf  = (float*)take((size_t)MROWS * D_MODEL * 4);
  float*  kf  = (float*)take((size_t)MROWS * KV_DIM * 4);
  float*  vf  = (float*)take((size_t)MROWS * KV_DIM * 4);
  __bf16* qb  = (__bf16*)take((size_t)MROWS * D_MODEL * 2);
  __bf16* kbb = (__bf16*)take((size_t)MROWS * KV_DIM * 2);
  __bf16* vbb = (__bf16*)take((size_t)MROWS * KV_DIM * 2);
  __bf16* ab  = (__bf16*)take((size_t)MROWS * D_MODEL * 2);

  auto cvt = [&](const float* src, __bf16* dst, long n) {
    f32_to_bf16_kernel<<<dim3((unsigned)((n + 255) / 256)), 256, 0, stream>>>(src, dst, n);
  };
  cvt(x,  xb,  (long)MROWS * D_MODEL);
  cvt(Wq, wqb, (long)D_MODEL * D_MODEL);
  cvt(Wk, wkb, (long)D_MODEL * KV_DIM);
  cvt(Wv, wvb, (long)D_MODEL * KV_DIM);
  cvt(Wo, wob, (long)D_MODEL * D_MODEL);

  gemm_bf16<<<dim3(MROWS / 128, D_MODEL / 64), 256, 0, stream>>>(xb, wqb, qf, MROWS, D_MODEL, D_MODEL);
  gemm_bf16<<<dim3(MROWS / 128, KV_DIM / 64),  256, 0, stream>>>(xb, wkb, kf, MROWS, KV_DIM, D_MODEL);
  gemm_bf16<<<dim3(MROWS / 128, KV_DIM / 64),  256, 0, stream>>>(xb, wvb, vf, MROWS, KV_DIM, D_MODEL);

  long nq = (long)MROWS * NUM_HEADS * 32;
  rope_bf16<<<dim3((unsigned)((nq + 255) / 256)), 256, 0, stream>>>(
      qf, cosT, sinT, qb, MROWS, NUM_HEADS, 0.125f);   // fold 1/sqrt(64) into Q
  long nk = (long)MROWS * NUM_KV * 32;
  rope_bf16<<<dim3((unsigned)((nk + 255) / 256)), 256, 0, stream>>>(
      kf, cosT, sinT, kbb, MROWS, NUM_KV, 1.0f);
  cvt(vf, vbb, (long)MROWS * KV_DIM);

  attn_fa<<<dim3(SEQ / 128, NUM_HEADS, BATCH), 256, 0, stream>>>(qb, kbb, vbb, ab);

  gemm_bf16<<<dim3(MROWS / 128, D_MODEL / 64), 256, 0, stream>>>(ab, wob, out, MROWS, D_MODEL, D_MODEL);

  (void)in_sizes; (void)n_in; (void)out_size; (void)ws_size;
}
